// CapsNet_17351667876147
// MI455X (gfx1250) — matrix operand order repaired and checked by reading, not compiled, not run on hardware
//
#include <hip/hip_runtime.h>
#include <math.h>

typedef __attribute__((ext_vector_type(16))) _Float16 v16h;
typedef __attribute__((ext_vector_type(8)))  _Float16 half8;
typedef __attribute__((ext_vector_type(8)))  float    v8f;

#define BATCH   512
#define C1      256      // conv1 out channels
#define H1      20
#define HW1     400      // 20*20
#define IMG     784      // 28*28
#define OC      256      // vconv out channels
#define OSP     36       // 6*6
#define KK      20736    // 256*81  (GEMM K)
#define MM      18432    // 512*36  (GEMM M)
#define KTILES  648      // 20736/32
#define P       1152
#define NK      10
#define E       16

// ---------------------------------------------------------------------------
// conv1 (1->256, 9x9, valid) + bias + relu, output fp16 NCHW
// ---------------------------------------------------------------------------
__global__ __launch_bounds__(256)
void conv1_relu_f16(const float* __restrict__ x, const float* __restrict__ w,
                    const float* __restrict__ bias, _Float16* __restrict__ h) {
  __shared__ float simg[IMG];
  const int b  = blockIdx.x;
  const int oc = threadIdx.x;
  for (int i = oc; i < IMG; i += 256) simg[i] = x[b * IMG + i];
  __syncthreads();
  float wreg[81];
#pragma unroll
  for (int j = 0; j < 81; ++j) wreg[j] = w[oc * 81 + j];
  const float bb = bias[oc];
  _Float16* hp = h + ((size_t)b * C1 + oc) * HW1;
  for (int y = 0; y < H1; ++y) {
    for (int xx = 0; xx < H1; ++xx) {
      float acc = bb;
#pragma unroll
      for (int ky = 0; ky < 9; ++ky)
#pragma unroll
        for (int kx = 0; kx < 9; ++kx)
          acc += simg[(y + ky) * 28 + (xx + kx)] * wreg[ky * 9 + kx];
      hp[y * H1 + xx] = (_Float16)fmaxf(acc, 0.0f);
    }
  }
}

// ---------------------------------------------------------------------------
// Repack vconv weights (256,256,9,9) fp32 -> fragment-ready f16 B tiles.
// Layout: [ktile 0..647][ntile 0..15][lane 0..31][i 0..15]  (16KB per ktile)
// K-permutation f folded in so the A side can be plain row-major:
//   k_local(lane,i) = ((i>>3)&1)*16 + (lane>>4)*8 + (i&7)
// ---------------------------------------------------------------------------
__global__ __launch_bounds__(256)
void pack_wb(const float* __restrict__ wconv, _Float16* __restrict__ wpack, int total) {
  int idx = blockIdx.x * 256 + threadIdx.x;
  if (idx >= total) return;
  int i    = idx & 15;
  int lane = (idx >> 4) & 31;
  int nt   = (idx >> 9) & 15;
  int t    = idx >> 13;
  int k = t * 32 + (((i >> 3) & 1) << 4) + ((lane >> 4) << 3) + (i & 7);
  int n = nt * 16 + (lane & 15);
  wpack[idx] = (_Float16)wconv[(size_t)n * KK + k];
}

// ---------------------------------------------------------------------------
// vconv implicit GEMM, v_wmma_f32_16x16x32_f16.
// Block = 256 thr = 8 waves; tile 64M x 256N; 8 WMMAs / wave / K-step.
// A panel (64x32, im2col) double-buffered in LDS (pitch 40 halves).
// B panel (16KB, fragment-packed) double-buffered in LDS via
// global_load_async_to_lds_b128 (ASYNCcnt), overlapped with compute.
// ---------------------------------------------------------------------------
__global__ __launch_bounds__(256)
void vconv_wmma(const _Float16* __restrict__ hbuf, const _Float16* __restrict__ wpack,
                const float* __restrict__ vbias, float* __restrict__ u_pre) {
  __shared__ _Float16 sA[2][64 * 40];        // 10 KB
  __shared__ _Float16 sB[2][16 * 32 * 16];   // 32 KB

  const int tid    = threadIdx.x;
  const int wid    = tid >> 5;
  const int lane   = tid & 31;
  const int wave_m = wid & 3;                // 0..3 -> 16 M rows each
  const int wave_n = wid >> 2;               // 0..1 -> 8 N tiles each
  const int Mbase  = blockIdx.x * 64;

  // per-thread im2col coords: 8 consecutive k per thread, fixed (row,col8)
  const int row  = tid >> 2;                 // 0..63
  const int col8 = (tid & 3) * 8;            // 0,8,16,24
  const int m    = Mbase + row;
  const int b    = m / OSP;
  const int s    = m - b * OSP;
  const int oy   = s / 6, ox = s - (s / 6) * 6;
  const _Float16* hb = hbuf + (size_t)b * (C1 * HW1) + (2 * oy) * H1 + (2 * ox);
  int ic = 0, r = col8;                      // k = ic*81 + r for current staged step

  const char* wp8 = (const char*)wpack;

  v8f acc[8];
#pragma unroll
  for (int q = 0; q < 8; ++q) acc[q] = (v8f){0.f,0.f,0.f,0.f,0.f,0.f,0.f,0.f};

  // ---- prologue: stage step 0 into buffer 0 ----
  {
    const char* gsrc = wp8;                  // step 0 B panel
#pragma unroll
    for (int j = 0; j < 4; ++j) {
      unsigned int loff =
          (unsigned int)(unsigned long long)(&sB[0][0]) + (unsigned)(j * 4096 + tid * 16);
      unsigned long long ga = (unsigned long long)(gsrc + j * 4096 + tid * 16);
      asm volatile("global_load_async_to_lds_b128 %0, %1, off"
                   :: "v"(loff), "v"(ga) : "memory");
    }
    half8 v;
    int icc = ic, rr = r;
#pragma unroll
    for (int j = 0; j < 8; ++j) {
      int ky = rr / 9, kx = rr - ky * 9;
      v[j] = hb[icc * HW1 + ky * H1 + kx];
      if (++rr >= 81) { rr = 0; ++icc; }
    }
    *(half8*)(&sA[0][row * 40 + col8]) = v;
  }

  for (int t = 0; t < KTILES; ++t) {
    const int cur = t & 1;
    // own async writes for buffer `cur` complete, then make all waves' LDS
    // writes visible (barrier also guarantees everyone is done READING the
    // buffer we are about to overwrite below).
    asm volatile("s_wait_asynccnt 0x0" ::: "memory");
    __syncthreads();

    if (t + 1 < KTILES) {
      const int nxt = cur ^ 1;
      const char* gsrc = wp8 + (size_t)(t + 1) * 16384;
#pragma unroll
      for (int j = 0; j < 4; ++j) {
        unsigned int loff =
            (unsigned int)(unsigned long long)(&sB[nxt][0]) + (unsigned)(j * 4096 + tid * 16);
        unsigned long long ga = (unsigned long long)(gsrc + j * 4096 + tid * 16);
        asm volatile("global_load_async_to_lds_b128 %0, %1, off"
                     :: "v"(loff), "v"(ga) : "memory");
      }
      // A panel for step t+1 (k advances by 32)
      int rn = r + 32, icn = ic;
      if (rn >= 81) { rn -= 81; ++icn; }
      half8 v;
      int icc = icn, rr = rn;
#pragma unroll
      for (int j = 0; j < 8; ++j) {
        int ky = rr / 9, kx = rr - ky * 9;
        v[j] = hb[icc * HW1 + ky * H1 + kx];
        if (++rr >= 81) { rr = 0; ++icc; }
      }
      *(half8*)(&sA[nxt][row * 40 + col8]) = v;
      r = rn; ic = icn;
    }

    // ---- compute step t from buffer `cur` ----
    union { v16h v; half8 h[2]; } Af;
    const _Float16* ap = &sA[cur][(wave_m * 16 + (lane & 15)) * 40 + (lane >> 4) * 8];
    Af.h[0] = *(const half8*)ap;
    Af.h[1] = *(const half8*)(ap + 16);
#pragma unroll
    for (int q = 0; q < 8; ++q) {
      union { v16h v; half8 h[2]; } Bf;
      const _Float16* bp = &sB[cur][((wave_n * 8 + q) * 32 + lane) * 16];
      Bf.h[0] = *(const half8*)bp;
      Bf.h[1] = *(const half8*)(bp + 8);
      acc[q] = __builtin_amdgcn_wmma_f32_16x16x32_f16(false, Af.v, false, Bf.v,
                                                      (short)0, acc[q], false, false);
    }
  }

  // epilogue: C layout lane l, vgpr j -> M = j + 8*(l>>4), N = l&15
  const int mrow0 = Mbase + wave_m * 16 + ((lane >> 4) * 8);
#pragma unroll
  for (int q = 0; q < 8; ++q) {
    const int ncol = (wave_n * 8 + q) * 16 + (lane & 15);
    const float bb = vbias[ncol];
#pragma unroll
    for (int j = 0; j < 8; ++j)
      u_pre[(size_t)(mrow0 + j) * OC + ncol] = acc[q][j] + bb;
  }
}

// ---------------------------------------------------------------------------
// squash over the 8-dim capsule axis: u_pre[m][oc] -> u[b][p][8]
// ---------------------------------------------------------------------------
__global__ __launch_bounds__(256)
void squash_u(const float* __restrict__ u_pre, float* __restrict__ u) {
  int g = blockIdx.x * 256 + threadIdx.x;
  if (g >= BATCH * P) return;
  int b = g / P, p = g - b * P;
  int c32 = p / OSP, s = p - c32 * OSP;
  const float* row = u_pre + (size_t)(b * OSP + s) * OC;
  float vals[8], n = 0.f;
#pragma unroll
  for (int d = 0; d < 8; ++d) { vals[d] = row[d * 32 + c32]; n += vals[d] * vals[d]; }
  float scale = n / ((n + 1.0f) * sqrtf(n));
  float* up = u + (size_t)g * 8;
#pragma unroll
  for (int d = 0; d < 8; ++d) up[d] = vals[d] * scale;
}

// ---------------------------------------------------------------------------
// u_hat[b,k,p,e] = sum_d u[b,p,d] * W[k,p,d,e]; one thread per (b,k,p)
// ---------------------------------------------------------------------------
__global__ __launch_bounds__(256)
void uhat_kernel(const float* __restrict__ u, const float* __restrict__ W,
                 float* __restrict__ u_hat) {
  int g = blockIdx.x * 256 + threadIdx.x;
  if (g >= BATCH * NK * P) return;
  int p = g % P;
  int k = (g / P) % NK;
  int b = g / (P * NK);
  const float* up = u + ((size_t)b * P + p) * 8;
  float ur[8];
#pragma unroll
  for (int d = 0; d < 8; ++d) ur[d] = up[d];
  float acc[E];
#pragma unroll
  for (int e = 0; e < E; ++e) acc[e] = 0.f;
  const float* wp = W + (((size_t)k * P + p) * 8) * E;
  for (int d = 0; d < 8; ++d) {
#pragma unroll
    for (int e = 0; e < E; ++e) acc[e] += ur[d] * wp[d * E + e];
  }
  float* op = u_hat + (((size_t)(b * NK + k)) * P + p) * E;
#pragma unroll
  for (int e = 0; e < E; ++e) op[e] = acc[e];
}

__global__ __launch_bounds__(256)
void zero_f32(float* __restrict__ ptr, int n) {
  int g = blockIdx.x * 256 + threadIdx.x;
  if (g < n) ptr[g] = 0.f;
}

// softmax over k for each (b,p)
__global__ __launch_bounds__(256)
void route_softmax(const float* __restrict__ Bs, float* __restrict__ Cs) {
  int g = blockIdx.x * 256 + threadIdx.x;
  if (g >= BATCH * P) return;
  int b = g / P, p = g - b * P;
  float t[NK], mx = -1e30f;
#pragma unroll
  for (int k = 0; k < NK; ++k) {
    t[k] = Bs[((size_t)(b * NK + k)) * P + p];
    mx = fmaxf(mx, t[k]);
  }
  float sum = 0.f;
#pragma unroll
  for (int k = 0; k < NK; ++k) { t[k] = expf(t[k] - mx); sum += t[k]; }
  float inv = 1.0f / sum;
#pragma unroll
  for (int k = 0; k < NK; ++k) Cs[((size_t)(b * NK + k)) * P + p] = t[k] * inv;
}

// s = sum_p Cs*u_hat, then v = squash(s); one block (128 thr) per (b,k)
__global__ __launch_bounds__(128)
void route_sv(const float* __restrict__ Cs, const float* __restrict__ u_hat,
              float* __restrict__ v) {
  __shared__ float red[128][E];
  const int b = blockIdx.x, k = blockIdx.y, tid = threadIdx.x;
  const float* uh = u_hat + ((size_t)(b * NK + k)) * P * E;
  const float* cp = Cs + ((size_t)(b * NK + k)) * P;
  float acc[E];
#pragma unroll
  for (int e = 0; e < E; ++e) acc[e] = 0.f;
  for (int p = tid; p < P; p += 128) {
    float c = cp[p];
    const float* u16 = uh + (size_t)p * E;
#pragma unroll
    for (int e = 0; e < E; ++e) acc[e] += c * u16[e];
  }
#pragma unroll
  for (int e = 0; e < E; ++e) red[tid][e] = acc[e];
  __syncthreads();
  for (int off = 64; off >= 1; off >>= 1) {
    if (tid < off) {
#pragma unroll
      for (int e = 0; e < E; ++e) red[tid][e] += red[tid + off][e];
    }
    __syncthreads();
  }
  if (tid == 0) {
    float n = 0.f;
#pragma unroll
    for (int e = 0; e < E; ++e) n += red[0][e] * red[0][e];
    float scale = n / ((n + 1.0f) * sqrtf(n));
    float* vp = v + ((size_t)(b * NK + k)) * E;
#pragma unroll
    for (int e = 0; e < E; ++e) vp[e] = red[0][e] * scale;
  }
}

// Bs += sum_e v*u_hat ; one thread per (b,k,p)
__global__ __launch_bounds__(256)
void route_update(const float* __restrict__ v, const float* __restrict__ u_hat,
                  float* __restrict__ Bs) {
  int g = blockIdx.x * 256 + threadIdx.x;
  if (g >= BATCH * NK * P) return;
  int p = g % P;
  int bk = g / P;
  const float* vp = v + (size_t)bk * E;
  const float* up = u_hat + ((size_t)bk * P + p) * E;
  float d = 0.f;
#pragma unroll
  for (int e = 0; e < E; ++e) d += vp[e] * up[e];
  Bs[(size_t)bk * P + p] += d;
}

// masked = v * one_hot(y)
__global__ __launch_bounds__(256)
void mask_kernel(const float* __restrict__ v, const int* __restrict__ y,
                 float* __restrict__ masked) {
  int g = blockIdx.x * 256 + threadIdx.x;
  if (g >= BATCH * 160) return;
  int b = g / 160, j = g - b * 160;
  int k = j >> 4;
  masked[g] = (y[b] == k) ? v[g] : 0.0f;
}

// out[b][n] = act(sum_k in[b][k]*w[k][n] + bias[n]); act 0=relu 1=sigmoid
__global__ __launch_bounds__(256)
void fc_kernel(const float* __restrict__ in, const float* __restrict__ w,
               const float* __restrict__ bias, float* __restrict__ out,
               int K, int N, int act) {
  extern __shared__ float sin_[];
  const int b = blockIdx.x;
  const int n = blockIdx.y * 256 + threadIdx.x;
  for (int i = threadIdx.x; i < K; i += 256) sin_[i] = in[(size_t)b * K + i];
  __syncthreads();
  if (n >= N) return;
  float acc = bias[n];
  for (int k = 0; k < K; ++k) acc += sin_[k] * w[(size_t)k * N + n];
  if (act == 0)      acc = fmaxf(acc, 0.0f);
  else               acc = 1.0f / (1.0f + expf(-acc));
  out[(size_t)b * N + n] = acc;
}

// ---------------------------------------------------------------------------
extern "C" void kernel_launch(void* const* d_in, const int* in_sizes, int n_in,
                              void* d_out, int out_size, void* d_ws, size_t ws_size,
                              hipStream_t stream) {
  const float* x       = (const float*)d_in[0];
  const int*   y       = (const int*)  d_in[1];
  const float* conv1_w = (const float*)d_in[2];
  const float* conv1_b = (const float*)d_in[3];
  const float* vconv_w = (const float*)d_in[4];
  const float* vconv_b = (const float*)d_in[5];
  const float* W       = (const float*)d_in[6];
  const float* w1      = (const float*)d_in[7];
  const float* b1      = (const float*)d_in[8];
  const float* w2      = (const float*)d_in[9];
  const float* b2      = (const float*)d_in[10];
  const float* w3      = (const float*)d_in[11];
  const float* b3      = (const float*)d_in[12];
  float* out = (float*)d_out;              // [v: 81920][recon: 401408]

  char* ws = (char*)d_ws;
  size_t off = 0;
  auto carve = [&](size_t bytes) {
    void* p = ws + off;
    off += (bytes + 255) & ~(size_t)255;
    return p;
  };
  _Float16* h16    = (_Float16*)carve((size_t)BATCH * C1 * HW1 * 2);      // 104.9 MB
  _Float16* wpack  = (_Float16*)carve((size_t)KK * OC * 2);               //  10.6 MB
  float*    u_pre  = (float*)   carve((size_t)MM * OC * 4);               //  18.9 MB
  float*    u      = (float*)   carve((size_t)BATCH * P * 8 * 4);         //  18.9 MB
  float*    u_hat  = (float*)   carve((size_t)BATCH * NK * P * E * 4);    // 377.5 MB
  float*    Bs     = (float*)   carve((size_t)BATCH * NK * P * 4);        //  23.6 MB
  float*    Cs     = (float*)   carve((size_t)BATCH * NK * P * 4);        //  23.6 MB
  float*    vbuf   = (float*)   carve((size_t)BATCH * NK * E * 4);
  float*    masked = (float*)   carve((size_t)BATCH * 160 * 4);
  float*    r1     = (float*)   carve((size_t)BATCH * 512 * 4);
  float*    r2     = (float*)   carve((size_t)BATCH * 1024 * 4);

  conv1_relu_f16<<<BATCH, 256, 0, stream>>>(x, conv1_w, conv1_b, h16);

  {
    int total = KTILES * 16 * 32 * 16;   // == KK * OC
    pack_wb<<<(total + 255) / 256, 256, 0, stream>>>(vconv_w, wpack, total);
  }

  vconv_wmma<<<MM / 64, 256, 0, stream>>>(h16, wpack, vconv_b, u_pre);

  squash_u<<<(BATCH * P + 255) / 256, 256, 0, stream>>>(u_pre, u);

  uhat_kernel<<<(BATCH * NK * P + 255) / 256, 256, 0, stream>>>(u, W, u_hat);

  zero_f32<<<(BATCH * NK * P + 255) / 256, 256, 0, stream>>>(Bs, BATCH * NK * P);
  for (int it = 0; it < 5; ++it) {
    route_softmax<<<(BATCH * P + 255) / 256, 256, 0, stream>>>(Bs, Cs);
    float* vtgt = (it == 4) ? out : vbuf;
    route_sv<<<dim3(BATCH, NK), 128, 0, stream>>>(Cs, u_hat, vtgt);
    if (it < 4)
      route_update<<<(BATCH * NK * P + 255) / 256, 256, 0, stream>>>(vbuf, u_hat, Bs);
  }

  mask_kernel<<<(BATCH * 160 + 255) / 256, 256, 0, stream>>>(out, y, masked);
  fc_kernel<<<dim3(BATCH, 2), 256, 160 * 4, stream>>>(masked, w1, b1, r1, 160, 512, 0);
  fc_kernel<<<dim3(BATCH, 4), 256, 512 * 4, stream>>>(r1, w2, b2, r2, 512, 1024, 0);
  fc_kernel<<<dim3(BATCH, 4), 256, 1024 * 4, stream>>>(r2, w3, b3, out + BATCH * NK * E,
                                                       1024, 784, 1);
}
